// GPT2Block_80436147519756
// MI455X (gfx1250) — matrix operand order repaired and checked by reading, not compiled, not run on hardware
//
#include <hip/hip_runtime.h>
#include <hip/hip_bf16.h>

typedef __attribute__((ext_vector_type(16))) __bf16 v16bf;
typedef __attribute__((ext_vector_type(8)))  __bf16 v8bf;
typedef __attribute__((ext_vector_type(8)))  float  v8f;
typedef unsigned short u16;
typedef unsigned int   u32;

// ---------- constants for this problem ----------
#define BB 4
#define SS 2048
#define DD 1024
#define FF 4096
#define HH 16
#define MM (BB*SS)          // 8192 token rows

__device__ __forceinline__ u16 f2bf(float f) {
    u32 u = __builtin_bit_cast(u32, f);
    u32 r = u + 0x7FFFu + ((u >> 16) & 1u);   // round-to-nearest-even
    return (u16)(r >> 16);
}
__device__ __forceinline__ v16bf cat8(v8bf lo, v8bf hi) {
    return __builtin_shufflevector(lo, hi, 0,1,2,3,4,5,6,7,8,9,10,11,12,13,14,15);
}
__device__ __forceinline__ float gelu_tanh(float x) {
    return 0.5f * x * (1.0f + tanhf(0.7978845608f * (x + 0.044715f*x*x*x)));
}

// ---------------- LayerNorm -> bf16 ----------------
__global__ __launch_bounds__(256)
void ln_bf16_kernel(const float* __restrict__ x, const float* __restrict__ g,
                    const float* __restrict__ b, u16* __restrict__ out) {
    int row = blockIdx.x;
    int t = threadIdx.x;
    const float* xr = x + (size_t)row * DD;
    float v[4]; float s = 0.f;
    #pragma unroll
    for (int i = 0; i < 4; ++i) { v[i] = xr[t + i*256]; s += v[i]; }
    __shared__ float red[256];
    red[t] = s; __syncthreads();
    for (int off = 128; off > 0; off >>= 1) { if (t < off) red[t] += red[t+off]; __syncthreads(); }
    float mu = red[0] * (1.0f/DD); __syncthreads();
    float s2 = 0.f;
    #pragma unroll
    for (int i = 0; i < 4; ++i) { float d = v[i]-mu; s2 += d*d; }
    red[t] = s2; __syncthreads();
    for (int off = 128; off > 0; off >>= 1) { if (t < off) red[t] += red[t+off]; __syncthreads(); }
    float rstd = rsqrtf(red[0] * (1.0f/DD) + 1e-5f);
    u16* orow = out + (size_t)row * DD;
    #pragma unroll
    for (int i = 0; i < 4; ++i) { int c = t + i*256; orow[c] = f2bf((v[i]-mu)*rstd*g[c] + b[c]); }
}

// ---------- tiled weight transpose + fp32 -> bf16 (out K-major), fully coalesced ----------
// in:  W[N][K] fp32 ; out: outT[k][colOff + n] bf16 with leading dim ldOut
__global__ __launch_bounds__(256)
void transpose_to_bf16(const float* __restrict__ W, u16* __restrict__ outT,
                       int N, int K, int ldOut, int colOff) {
    __shared__ u16 tile[32][33];
    int k0 = blockIdx.x * 32, n0 = blockIdx.y * 32;
    int tx = threadIdx.x & 31, ty = threadIdx.x >> 5;   // 32 x 8
    #pragma unroll
    for (int i = 0; i < 4; ++i) {
        int n = n0 + ty + i*8;
        tile[ty + i*8][tx] = f2bf(W[(size_t)n * K + k0 + tx]);   // coalesced over tx (k)
    }
    __syncthreads();
    #pragma unroll
    for (int i = 0; i < 4; ++i) {
        int k = k0 + ty + i*8;
        outT[(size_t)k * ldOut + colOff + n0 + tx] = tile[tx][ty + i*8];  // coalesced over tx (n)
    }
}

// ---------------- 128x128 bf16 WMMA GEMM, 256 threads / 8 waves, K-step 64 ----------------
// MODE 0: outF[row,col] = acc + bias + resid[row,col]          (fp32 store)
// MODE 1: outB[row,col] = bf16( gelu(acc + bias) )             (bf16 store)
template <int MODE>
__global__ __launch_bounds__(256)
void gemm_bf16(const u16* __restrict__ A, const u16* __restrict__ Bt,
               const float* __restrict__ bias, int N, int K,
               const float* __restrict__ resid, float* __restrict__ outF,
               u16* __restrict__ outB)
{
    __shared__ __attribute__((aligned(16))) u16 As[128][72];   // 128 x 64 (+8 pad)
    __shared__ __attribute__((aligned(16))) u16 Bs[64][136];   //  64 x128 (+8 pad)
    int t = threadIdx.x, lane = t & 31, wave = t >> 5;
    int mBlock = blockIdx.y * 128, nBlock = blockIdx.x * 128;
    int wm = (wave & 3) * 32, wn = (wave >> 2) * 64;           // wave tile: 32 x 64
    int hi8 = (lane < 16) ? 0 : 8;
    v8f acc[2][4] = {};

    int ar_ = t >> 1, ac_ = (t & 1) * 32;                      // A staging coords
    int br_ = t >> 2, bc_ = (t & 3) * 32;                      // B staging coords
    const u16* aSrc = A  + (size_t)(mBlock + ar_) * K + ac_;
    const u16* bSrc = Bt + (size_t)br_ * N + nBlock + bc_;

    for (int kb = 0; kb < K; kb += 64) {
        { // stage A 128x64
            const uint4* s4 = (const uint4*)(aSrc + kb);
            uint4 a0 = s4[0], a1 = s4[1], a2 = s4[2], a3 = s4[3];
            *(uint4*)&As[ar_][ac_]      = a0;
            *(uint4*)&As[ar_][ac_ + 8]  = a1;
            *(uint4*)&As[ar_][ac_ + 16] = a2;
            *(uint4*)&As[ar_][ac_ + 24] = a3;
        }
        { // stage B 64x128
            const uint4* s4 = (const uint4*)(bSrc + (size_t)kb * N);
            uint4 b0 = s4[0], b1 = s4[1], b2 = s4[2], b3 = s4[3];
            *(uint4*)&Bs[br_][bc_]      = b0;
            *(uint4*)&Bs[br_][bc_ + 8]  = b1;
            *(uint4*)&Bs[br_][bc_ + 16] = b2;
            *(uint4*)&Bs[br_][bc_ + 24] = b3;
        }
        if (kb + 64 < K) {  // prefetch next K tile (global_prefetch_b8)
            __builtin_prefetch(aSrc + kb + 64, 0, 1);
            __builtin_prefetch(bSrc + (size_t)(kb + 64) * N, 0, 1);
        }
        __syncthreads();
        #pragma unroll
        for (int ks = 0; ks < 64; ks += 32) {
            v16bf af[2], bfr[4];
            #pragma unroll
            for (int i = 0; i < 2; ++i) {
                const u16* ar = &As[wm + i*16 + (lane & 15)][ks + hi8];
                af[i] = cat8(*(const v8bf*)ar, *(const v8bf*)(ar + 16));
            }
            #pragma unroll
            for (int j = 0; j < 4; ++j)
                for (int s = 0; s < 16; ++s)
                    bfr[j][s] = __builtin_bit_cast(__bf16, Bs[ks + hi8 + (s & 7) + ((s >> 3) << 4)][wn + j*16 + (lane & 15)]);
            #pragma unroll
            for (int i = 0; i < 2; ++i)
                #pragma unroll
                for (int j = 0; j < 4; ++j)
                    acc[i][j] = __builtin_amdgcn_wmma_f32_16x16x32_bf16(
                        false, af[i], false, bfr[j], (short)0, acc[i][j], false, false);
        }
        __syncthreads();
    }
    #pragma unroll
    for (int i = 0; i < 2; ++i)
        #pragma unroll
        for (int j = 0; j < 4; ++j) {
            int col = nBlock + wn + j*16 + (lane & 15);
            float bcol = bias[col];
            #pragma unroll
            for (int r = 0; r < 8; ++r) {
                int row = mBlock + wm + i*16 + r + hi8;
                float val = acc[i][j][r] + bcol;
                if (MODE == 0) {
                    val += resid[(size_t)row * N + col];
                    outF[(size_t)row * N + col] = val;
                } else {
                    outB[(size_t)row * N + col] = f2bf(gelu_tanh(val));
                }
            }
        }
}

// ---------------- QKV GEMM 128x128: writes bf16 Q(scaled)/K/V in (B,H,S,Dh) ----------------
__global__ __launch_bounds__(256)
void gemm_qkv(const u16* __restrict__ A, const u16* __restrict__ Bt,
              const float* __restrict__ bias,
              u16* __restrict__ Qo, u16* __restrict__ Ko, u16* __restrict__ Vo)
{
    const int N = 3 * DD, K = DD;
    __shared__ __attribute__((aligned(16))) u16 As[128][72];
    __shared__ __attribute__((aligned(16))) u16 Bs[64][136];
    int t = threadIdx.x, lane = t & 31, wave = t >> 5;
    int mBlock = blockIdx.y * 128, nBlock = blockIdx.x * 128;
    int wm = (wave & 3) * 32, wn = (wave >> 2) * 64;
    int hi8 = (lane < 16) ? 0 : 8;
    v8f acc[2][4] = {};

    int ar_ = t >> 1, ac_ = (t & 1) * 32;
    int br_ = t >> 2, bc_ = (t & 3) * 32;
    const u16* aSrc = A  + (size_t)(mBlock + ar_) * K + ac_;
    const u16* bSrc = Bt + (size_t)br_ * N + nBlock + bc_;

    for (int kb = 0; kb < K; kb += 64) {
        {
            const uint4* s4 = (const uint4*)(aSrc + kb);
            uint4 a0 = s4[0], a1 = s4[1], a2 = s4[2], a3 = s4[3];
            *(uint4*)&As[ar_][ac_]      = a0;
            *(uint4*)&As[ar_][ac_ + 8]  = a1;
            *(uint4*)&As[ar_][ac_ + 16] = a2;
            *(uint4*)&As[ar_][ac_ + 24] = a3;
        }
        {
            const uint4* s4 = (const uint4*)(bSrc + (size_t)kb * N);
            uint4 b0 = s4[0], b1 = s4[1], b2 = s4[2], b3 = s4[3];
            *(uint4*)&Bs[br_][bc_]      = b0;
            *(uint4*)&Bs[br_][bc_ + 8]  = b1;
            *(uint4*)&Bs[br_][bc_ + 16] = b2;
            *(uint4*)&Bs[br_][bc_ + 24] = b3;
        }
        if (kb + 64 < K) {
            __builtin_prefetch(aSrc + kb + 64, 0, 1);
            __builtin_prefetch(bSrc + (size_t)(kb + 64) * N, 0, 1);
        }
        __syncthreads();
        #pragma unroll
        for (int ks = 0; ks < 64; ks += 32) {
            v16bf af[2], bfr[4];
            #pragma unroll
            for (int i = 0; i < 2; ++i) {
                const u16* ar = &As[wm + i*16 + (lane & 15)][ks + hi8];
                af[i] = cat8(*(const v8bf*)ar, *(const v8bf*)(ar + 16));
            }
            #pragma unroll
            for (int j = 0; j < 4; ++j)
                for (int s = 0; s < 16; ++s)
                    bfr[j][s] = __builtin_bit_cast(__bf16, Bs[ks + hi8 + (s & 7) + ((s >> 3) << 4)][wn + j*16 + (lane & 15)]);
            #pragma unroll
            for (int i = 0; i < 2; ++i)
                #pragma unroll
                for (int j = 0; j < 4; ++j)
                    acc[i][j] = __builtin_amdgcn_wmma_f32_16x16x32_bf16(
                        false, af[i], false, bfr[j], (short)0, acc[i][j], false, false);
        }
        __syncthreads();
    }
    #pragma unroll
    for (int i = 0; i < 2; ++i)
        #pragma unroll
        for (int j = 0; j < 4; ++j) {
            int n = nBlock + wn + j*16 + (lane & 15);
            float bcol = bias[n];
            int part = n >> 10, f = n & 1023;
            int head = f >> 6, d = f & 63;
            u16* dst = (part == 0) ? Qo : ((part == 1) ? Ko : Vo);
            float scale = (part == 0) ? 0.125f : 1.0f;   // 1/sqrt(Dh=64)
            #pragma unroll
            for (int r = 0; r < 8; ++r) {
                int row = mBlock + wm + i*16 + r + hi8;
                int b = row >> 11, srow = row & 2047;
                float val = (acc[i][j][r] + bcol) * scale;
                dst[(((size_t)(b*HH + head)) * SS + srow) * 64 + d] = f2bf(val);
            }
        }
}

// ---------------- causal flash attention, WMMA, 4 waves x 16 q-rows ----------------
__global__ __launch_bounds__(128)
void attn_kernel(const u16* __restrict__ Qb, const u16* __restrict__ Kb,
                 const u16* __restrict__ Vb, u16* __restrict__ Ob)
{
    __shared__ __attribute__((aligned(16))) u16 Kt[64][72];      // [d][kv]  (K transposed)
    __shared__ __attribute__((aligned(16))) u16 Vs[64][72];      // [kv][d]
    __shared__ __attribute__((aligned(16))) u16 Ps[4][16][72];   // per-wave P tile
    int t = threadIdx.x, lane = t & 31, wave = t >> 5;
    int bh = blockIdx.y;                  // b*16 + h
    int b = bh >> 4, h = bh & 15;
    int qBlock = blockIdx.x * 64;
    int qRow0 = qBlock + wave * 16;
    int hi8 = (lane < 16) ? 0 : 8;

    const u16* qbase = Qb + ((size_t)bh * SS + qRow0 + (lane & 15)) * 64;
    v16bf qf[2];
    #pragma unroll
    for (int ks = 0; ks < 2; ++ks) {
        int cc = ks*32 + hi8;
        qf[ks] = cat8(*(const v8bf*)(qbase + cc), *(const v8bf*)(qbase + cc + 16));
    }
    v8f o[4] = {};
    float mrow[8], lrow[8];
    #pragma unroll
    for (int r = 0; r < 8; ++r) { mrow[r] = -INFINITY; lrow[r] = 0.f; }

    int nTiles = qBlock/64 + 1;
    for (int kt = 0; kt < nTiles; ++kt) {
        int kvBase = kt * 64;
        { // stage K (transposed) and V
            int r = t >> 1, c = (t & 1) * 32;
            const u16* ksrc = Kb + ((size_t)bh * SS + kvBase + r) * 64 + c;
            const u16* vsrc = Vb + ((size_t)bh * SS + kvBase + r) * 64 + c;
            u16 tmp[32];
            *(uint4*)&tmp[0]  = ((const uint4*)ksrc)[0];
            *(uint4*)&tmp[8]  = ((const uint4*)ksrc)[1];
            *(uint4*)&tmp[16] = ((const uint4*)ksrc)[2];
            *(uint4*)&tmp[24] = ((const uint4*)ksrc)[3];
            #pragma unroll
            for (int i = 0; i < 32; ++i) Kt[c + i][r] = tmp[i];
            *(uint4*)&Vs[r][c]      = ((const uint4*)vsrc)[0];
            *(uint4*)&Vs[r][c + 8]  = ((const uint4*)vsrc)[1];
            *(uint4*)&Vs[r][c + 16] = ((const uint4*)vsrc)[2];
            *(uint4*)&Vs[r][c + 24] = ((const uint4*)vsrc)[3];
        }
        __syncthreads();
        // S = Q @ K^T
        v8f sacc[4] = {};
        #pragma unroll
        for (int j = 0; j < 4; ++j)
            #pragma unroll
            for (int ks = 0; ks < 2; ++ks) {
                v16bf bk;
                for (int s = 0; s < 16; ++s)
                    bk[s] = __builtin_bit_cast(__bf16, Kt[ks*32 + hi8 + (s & 7) + ((s >> 3) << 4)][j*16 + (lane & 15)]);
                sacc[j] = __builtin_amdgcn_wmma_f32_16x16x32_bf16(
                    false, qf[ks], false, bk, (short)0, sacc[j], false, false);
            }
        // causal mask + online softmax
        #pragma unroll
        for (int r = 0; r < 8; ++r) {
            int qAbs = qRow0 + hi8 + r;
            float vmax = -3.0e38f;
            #pragma unroll
            for (int j = 0; j < 4; ++j) {
                int kvAbs = kvBase + j*16 + (lane & 15);
                float sv = sacc[j][r];
                if (kvAbs > qAbs) sv = -3.0e38f;
                sacc[j][r] = sv;
                vmax = fmaxf(vmax, sv);
            }
            #pragma unroll
            for (int off = 1; off < 16; off <<= 1) vmax = fmaxf(vmax, __shfl_xor(vmax, off, 32));
            float mnew  = fmaxf(mrow[r], vmax);
            float scale = __expf(mrow[r] - mnew);
            mrow[r] = mnew;
            float rsum = 0.f;
            #pragma unroll
            for (int j = 0; j < 4; ++j) {
                float p = __expf(sacc[j][r] - mnew);
                sacc[j][r] = p;
                rsum += p;
            }
            #pragma unroll
            for (int off = 1; off < 16; off <<= 1) rsum += __shfl_xor(rsum, off, 32);
            lrow[r] = lrow[r] * scale + rsum;
            #pragma unroll
            for (int j = 0; j < 4; ++j) o[j][r] *= scale;
        }
        // restage P (C-layout) into A-fragment layout via LDS
        #pragma unroll
        for (int r = 0; r < 8; ++r) {
            int rr = r + hi8;
            #pragma unroll
            for (int j = 0; j < 4; ++j)
                Ps[wave][rr][j*16 + (lane & 15)] = f2bf(sacc[j][r]);
        }
        __syncthreads();
        // O += P @ V
        #pragma unroll
        for (int j = 0; j < 4; ++j)
            #pragma unroll
            for (int ks = 0; ks < 2; ++ks) {
                const u16* pr = &Ps[wave][lane & 15][ks*32 + hi8];
                v16bf af = cat8(*(const v8bf*)pr, *(const v8bf*)(pr + 16));
                v16bf bv;
                for (int s = 0; s < 16; ++s)
                    bv[s] = __builtin_bit_cast(__bf16, Vs[ks*32 + hi8 + (s & 7) + ((s >> 3) << 4)][j*16 + (lane & 15)]);
                o[j] = __builtin_amdgcn_wmma_f32_16x16x32_bf16(
                    false, af, false, bv, (short)0, o[j], false, false);
            }
        __syncthreads();
    }
    #pragma unroll
    for (int j = 0; j < 4; ++j)
        #pragma unroll
        for (int r = 0; r < 8; ++r) {
            int qAbs = qRow0 + hi8 + r;
            float val = o[j][r] / lrow[r];
            size_t idx = ((size_t)b * SS + qAbs) * (size_t)DD + h*64 + j*16 + (lane & 15);
            Ob[idx] = f2bf(val);
        }
}

// ---------------------------------------------------------------------------
extern "C" void kernel_launch(void* const* d_in, const int* in_sizes, int n_in,
                              void* d_out, int out_size, void* d_ws, size_t ws_size,
                              hipStream_t stream) {
    const float* h      = (const float*)d_in[0];
    const float* ln1_g  = (const float*)d_in[1];
    const float* ln1_b  = (const float*)d_in[2];
    const float* wq     = (const float*)d_in[3];
    const float* bq     = (const float*)d_in[4];
    const float* wk     = (const float*)d_in[5];
    const float* bk     = (const float*)d_in[6];
    const float* wv     = (const float*)d_in[7];
    const float* bv     = (const float*)d_in[8];
    const float* wo     = (const float*)d_in[9];
    const float* bo     = (const float*)d_in[10];
    const float* ln2_g  = (const float*)d_in[11];
    const float* ln2_b  = (const float*)d_in[12];
    const float* wfc    = (const float*)d_in[13];
    const float* bfc    = (const float*)d_in[14];
    const float* wproj  = (const float*)d_in[15];
    const float* bproj  = (const float*)d_in[16];
    float* out = (float*)d_out;

    char* ws = (char*)d_ws;
    size_t off = 0;
    auto alloc = [&](size_t bytes) -> char* {
        char* p = ws + off; off += (bytes + 255) & ~(size_t)255; return p;
    };
    u16*   wqkvT  = (u16*)  alloc((size_t)DD * 3*DD * 2);   // 6 MB   [K=1024][N=3072]
    u16*   woT    = (u16*)  alloc((size_t)DD * DD   * 2);   // 2 MB
    u16*   wfcT   = (u16*)  alloc((size_t)DD * FF   * 2);   // 8 MB   [1024][4096]
    u16*   wprojT = (u16*)  alloc((size_t)FF * DD   * 2);   // 8 MB   [4096][1024]
    float* bqkv   = (float*)alloc(3*DD * 4);
    u16*   xln    = (u16*)  alloc((size_t)MM * DD * 2);     // 16 MB (LN1 then LN2)
    char*  scratch= alloc((size_t)MM * FF * 2);             // 64 MB: Q/K/V/attn, later a_fc
    float* h2     = (float*)alloc((size_t)MM * DD * 4);     // 32 MB

    u16* Qb   = (u16*)(scratch);
    u16* Kb   = (u16*)(scratch + (size_t)MM * DD * 2);
    u16* Vb   = (u16*)(scratch + 2*(size_t)MM * DD * 2);
    u16* attn = (u16*)(scratch + 3*(size_t)MM * DD * 2);
    u16* afc  = (u16*)(scratch);                            // overlays Q/K/V/attn (dead by then)

    // --- weight prep: coalesced tiled transpose to K-major bf16 ---
    transpose_to_bf16<<<dim3(DD/32, DD/32), 256, 0, stream>>>(wq,    wqkvT,  DD, DD, 3*DD, 0);
    transpose_to_bf16<<<dim3(DD/32, DD/32), 256, 0, stream>>>(wk,    wqkvT,  DD, DD, 3*DD, DD);
    transpose_to_bf16<<<dim3(DD/32, DD/32), 256, 0, stream>>>(wv,    wqkvT,  DD, DD, 3*DD, 2*DD);
    transpose_to_bf16<<<dim3(DD/32, DD/32), 256, 0, stream>>>(wo,    woT,    DD, DD, DD,   0);
    transpose_to_bf16<<<dim3(DD/32, FF/32), 256, 0, stream>>>(wfc,   wfcT,   FF, DD, FF,   0);
    transpose_to_bf16<<<dim3(FF/32, DD/32), 256, 0, stream>>>(wproj, wprojT, DD, FF, DD,   0);
    hipMemcpyAsync(bqkv,        bq, DD*4, hipMemcpyDeviceToDevice, stream);
    hipMemcpyAsync(bqkv + DD,   bk, DD*4, hipMemcpyDeviceToDevice, stream);
    hipMemcpyAsync(bqkv + 2*DD, bv, DD*4, hipMemcpyDeviceToDevice, stream);

    // --- attention sub-block ---
    ln_bf16_kernel<<<MM, 256, 0, stream>>>(h, ln1_g, ln1_b, xln);
    gemm_qkv<<<dim3(3*DD/128, MM/128), 256, 0, stream>>>(xln, wqkvT, bqkv, Qb, Kb, Vb);
    attn_kernel<<<dim3(SS/64, BB*HH), 128, 0, stream>>>(Qb, Kb, Vb, attn);
    gemm_bf16<0><<<dim3(DD/128, MM/128), 256, 0, stream>>>(
        attn, woT, bo, DD, DD, /*resid=*/h, /*outF=*/h2, nullptr);

    // --- MLP sub-block ---
    ln_bf16_kernel<<<MM, 256, 0, stream>>>(h2, ln2_g, ln2_b, xln);
    gemm_bf16<1><<<dim3(FF/128, MM/128), 256, 0, stream>>>(
        xln, wfcT, bfc, FF, DD, nullptr, nullptr, /*outB=*/afc);
    gemm_bf16<0><<<dim3(DD/128, MM/128), 256, 0, stream>>>(
        afc, wprojT, bproj, DD, FF, /*resid=*/h2, /*outF=*/out, nullptr);

    (void)in_sizes; (void)n_in; (void)out_size; (void)ws_size;
}
